// Bsl2_9053791060551
// MI455X (gfx1250) — compile-verified, matrix operand order checked by reading, not run on hardware
//
#include <hip/hip_runtime.h>
#include <hip/hip_bf16.h>
#include <math.h>

// ---------------------------------------------------------------------------
// Problem constants
// ---------------------------------------------------------------------------
#define S_LEN 1024
#define B_SZ  64
#define I_DIM 512
#define H_DIM 256
#define MLP_D 512
#define K_SEC 64
#define G3H   (3 * H_DIM)     // 768
#define ROWS  (S_LEN * B_SZ)  // 65536
#define D2H   (2 * H_DIM)     // 512
#define D4H   (4 * H_DIM)     // 1024

typedef __bf16 bf16;
typedef __attribute__((ext_vector_type(16))) __bf16 v16bf;
typedef __attribute__((ext_vector_type(8)))  __bf16 v8bf;
typedef __attribute__((ext_vector_type(4)))  __bf16 v4bf;
typedef __attribute__((ext_vector_type(8)))  float  v8f;
typedef __attribute__((ext_vector_type(4)))  float  v4f;
typedef __attribute__((ext_vector_type(4)))  int    v4i;

union frag_u { v16bf v; v8bf h[2]; };

#define AS1 __attribute__((address_space(1)))
#define AS3 __attribute__((address_space(3)))

// Async global->LDS copy (CDNA5 GLOBAL_LOAD_ASYNC_TO_LDS_B128, ASYNCcnt).
#if defined(__has_builtin)
#  if __has_builtin(__builtin_amdgcn_global_load_async_to_lds_b128)
#    define HAVE_ASYNC_LDS 1
#  endif
#endif
#ifndef HAVE_ASYNC_LDS
#  define HAVE_ASYNC_LDS 0
#endif

__device__ __forceinline__ void async_copy16(bf16* lds_dst, const bf16* gsrc) {
#if HAVE_ASYNC_LDS
    __builtin_amdgcn_global_load_async_to_lds_b128(
        (AS1 v4i*)gsrc, (AS3 v4i*)lds_dst, 0, 0);
#else
    *(v8bf*)lds_dst = *(const v8bf*)gsrc;
#endif
}
__device__ __forceinline__ void async_wait_all() {
#if HAVE_ASYNC_LDS
    asm volatile("s_wait_asynccnt 0" ::: "memory");
#endif
}

// ---------------------------------------------------------------------------
// WMMA helpers (CDNA5 wave32: V_WMMA_F32_16X16X32_BF16)
//
// A fragment (16x32, MxK):  lane l: half=l>>4, m=l&15.
//   elements 0..7  = A[m, k0 + half*8 + 0..7]
//   elements 8..15 = A[m, k0 + half*8 + 16..23]
// B fragment (32x16, KxN):  same with n = l&15 over W rows (W is [N,K] row-major,
//   so B[k,n] = W[n,k] and the per-lane reads are contiguous in k).
// C/D (16x16 f32): lane l holds column n=l&15; vgpr r holds row m = r + 8*(l>>4).
// ---------------------------------------------------------------------------
__device__ __forceinline__ v16bf load_frag(const bf16* p, int ld, int rcbase, int k) {
    const int lane = threadIdx.x & 31;
    const int half = lane >> 4;
    const int idx  = lane & 15;
    const bf16* q = p + (size_t)(rcbase + idx) * (size_t)ld + k + half * 8;
    frag_u u;
    u.h[0] = *(const v8bf*)(q);
    u.h[1] = *(const v8bf*)(q + 16);
    return u.v;
}

__device__ __forceinline__ v8f wmma_bf16(v16bf a, v16bf b, v8f c) {
    return __builtin_amdgcn_wmma_f32_16x16x32_bf16(
        /*neg_a=*/false, a, /*neg_b=*/false, b,
        /*c_mod=*/(short)0, c, /*reuse_a=*/false, /*reuse_b=*/false);
}

__device__ __forceinline__ float sigmoidf_(float x) {
    return 1.0f / (1.0f + __expf(-x));
}

#define ZERO8 {0.f,0.f,0.f,0.f,0.f,0.f,0.f,0.f}

// ---------------------------------------------------------------------------
// Kernel: fp32 -> bf16 conversion, 4-wide (n must be divisible by 4)
// ---------------------------------------------------------------------------
__global__ void k_f32_to_bf16(const float* __restrict__ in, bf16* __restrict__ out, size_t n4) {
    size_t i = (size_t)blockIdx.x * blockDim.x + threadIdx.x;
    size_t stride = (size_t)gridDim.x * blockDim.x;
    for (; i < n4; i += stride) {
        v4f x = ((const v4f*)in)[i];
        ((v4bf*)out)[i] = __builtin_convertvector(x, v4bf);
    }
}

// Zero pad_output row t==0 (po is bf16 [B][S+1][2H])
__global__ void k_zero_po_row0(bf16* __restrict__ po) {
    int i = blockIdx.x * blockDim.x + threadIdx.x;
    if (i < B_SZ * D2H) {
        int b = i >> 9, c = i & (D2H - 1);
        po[(size_t)b * (S_LEN + 1) * D2H + c] = (bf16)0.f;
    }
}

// ---------------------------------------------------------------------------
// Kernel: xi = x @ W_ih^T + b_ih, stored bf16.
//   A: bf16 [M=65536, K=512] (x), W: bf16 [N=768, K=512], C: bf16 [M, N]
// Block = 256 threads (8 waves). Block tile = 128 rows x 64 cols; each wave
// owns 16 rows x 64 cols (4 accumulators), so one A fragment feeds 4 WMMAs.
// W is staged in 16 KB LDS chunks (64 cols x 128 k) via async-to-LDS.
// ---------------------------------------------------------------------------
__global__ __launch_bounds__(256) void k_gemm_xi(const bf16* __restrict__ A,
                                                 const bf16* __restrict__ W,
                                                 const float* __restrict__ bias,
                                                 bf16* __restrict__ C) {
    __shared__ bf16 Bs[64 * 128];  // 16 KB chunk

    const int wave = threadIdx.x >> 5;
    const int lane = threadIdx.x & 31;
    const int half = lane >> 4, nl = lane & 15;
    const int rbase = blockIdx.y * 128 + wave * 16;
    const int cblk  = blockIdx.x * 64;

    v8f acc[4] = {ZERO8, ZERO8, ZERO8, ZERO8};
    for (int kc = 0; kc < I_DIM; kc += 128) {
        // stage W rows [cblk, cblk+64) x k [kc, kc+128)
        for (int c = threadIdx.x; c < 64 * 128 / 8; c += 256) {
            int rr = c >> 4;              // 16 chunks of 8 bf16 per row
            int kk = (c & 15) << 3;
            async_copy16(Bs + rr * 128 + kk,
                         W + (size_t)(cblk + rr) * I_DIM + kc + kk);
        }
        async_wait_all();
        __syncthreads();
        #pragma unroll
        for (int k = 0; k < 128; k += 32) {
            v16bf a = load_frag(A, I_DIM, rbase, kc + k);
            #pragma unroll
            for (int ct = 0; ct < 4; ++ct)
                acc[ct] = wmma_bf16(a, load_frag(Bs, 128, ct * 16, k), acc[ct]);
        }
        __syncthreads();
    }
    #pragma unroll
    for (int ct = 0; ct < 4; ++ct) {
        const int cb = cblk + ct * 16;
        const float bv = bias[cb + nl];
        #pragma unroll
        for (int r = 0; r < 8; ++r) {
            int m = rbase + r + 8 * half;
            C[(size_t)m * G3H + cb + nl] = (bf16)(acc[ct][r] + bv);
        }
    }
}

// ---------------------------------------------------------------------------
// Kernel: GRU scan (persistent). grid = (2 dirs, 4 batch-tiles), block = 256.
// Each block owns 16 batch rows; h tile (16 x 256, bf16) lives in LDS.
// Per step: one h-fragment per k-step feeds 6 WMMAs (2 column units x 3
// gates); next step's 24 KB xi slab is prefetched (global_prefetch_b8) while
// the WMMAs run. W_hh stays L2-resident (384 KB).
// ---------------------------------------------------------------------------
__global__ __launch_bounds__(256) void k_gru_scan(const bf16* __restrict__ xi_f,
                                                  const bf16* __restrict__ xi_b,
                                                  const bf16* __restrict__ Whf,
                                                  const bf16* __restrict__ Whb,
                                                  const float* __restrict__ bhhf,
                                                  const float* __restrict__ bhhb,
                                                  bf16* __restrict__ po,
                                                  bf16* __restrict__ mlp_in) {
    __shared__ bf16 hl[16 * H_DIM];  // 8 KB

    const int d   = blockIdx.x;            // direction
    const int rb  = blockIdx.y * 16;       // batch-row base
    const bf16*  xi  = d ? xi_b  : xi_f;
    const bf16*  Wh  = d ? Whb   : Whf;
    const float* bhh = d ? bhhb  : bhhf;

    for (int i = threadIdx.x; i < 16 * H_DIM; i += 256) hl[i] = (bf16)0.f;
    __syncthreads();

    const int wave = threadIdx.x >> 5;
    const int lane = threadIdx.x & 31;
    const int half = lane >> 4, nl = lane & 15;

    // prefetch first step's xi slab
    {
        const int s0 = d ? (S_LEN - 1) : 0;
        const char* base = (const char*)(xi + ((size_t)s0 * B_SZ + rb) * G3H);
        int off = threadIdx.x * 128;
        if (off < 16 * G3H * 2) __builtin_prefetch(base + off, 0, 3);
    }

    for (int t = 0; t < S_LEN; ++t) {
        const int s = d ? (S_LEN - 1 - t) : t;

        // prefetch next step's xi slab so it lands while WMMAs run
        if (t + 1 < S_LEN) {
            const int sn = d ? (S_LEN - 2 - t) : (t + 1);
            const char* base = (const char*)(xi + ((size_t)sn * B_SZ + rb) * G3H);
            int off = threadIdx.x * 128;
            if (off < 16 * G3H * 2) __builtin_prefetch(base + off, 0, 3);
        }

        v8f acc[2][3] = {{ZERO8, ZERO8, ZERO8}, {ZERO8, ZERO8, ZERO8}};
        #pragma unroll
        for (int k = 0; k < H_DIM; k += 32) {
            v16bf a = load_frag(hl, H_DIM, 0, k);
            #pragma unroll
            for (int u = 0; u < 2; ++u) {
                const int j = wave * 2 + u;     // h-column tile 0..15
                acc[u][0] = wmma_bf16(a, load_frag(Wh, H_DIM, j * 16,             k), acc[u][0]);
                acc[u][1] = wmma_bf16(a, load_frag(Wh, H_DIM, H_DIM     + j * 16, k), acc[u][1]);
                acc[u][2] = wmma_bf16(a, load_frag(Wh, H_DIM, 2 * H_DIM + j * 16, k), acc[u][2]);
            }
        }

        float hnew[2][8];
        #pragma unroll
        for (int u = 0; u < 2; ++u) {
            const int j = wave * 2 + u;
            const int hidx = j * 16 + nl;
            const float br = bhh[hidx], bz = bhh[H_DIM + hidx], bn = bhh[2 * H_DIM + hidx];
            #pragma unroll
            for (int r = 0; r < 8; ++r) {
                const int bl = r + 8 * half;        // local batch row 0..15
                const int b  = rb + bl;
                const size_t row = (size_t)s * B_SZ + b;
                const float xr = (float)xi[row * G3H + hidx];
                const float xz = (float)xi[row * G3H + H_DIM + hidx];
                const float xn = (float)xi[row * G3H + 2 * H_DIM + hidx];
                const float rr = sigmoidf_(xr + acc[u][0][r] + br);
                const float zz = sigmoidf_(xz + acc[u][1][r] + bz);
                const float nn = tanhf(xn + rr * (acc[u][2][r] + bn));
                const float hold = (float)hl[bl * H_DIM + hidx];
                const float hv = (1.f - zz) * nn + zz * hold;
                hnew[u][r] = hv;
                po[((size_t)b * (S_LEN + 1) + (s + 1)) * D2H + d * H_DIM + hidx] = (bf16)hv;
                mlp_in[row * D4H + d * H_DIM + hidx] = (bf16)hv;
            }
        }
        __syncthreads();   // all reads of hl done
        #pragma unroll
        for (int u = 0; u < 2; ++u) {
            const int j = wave * 2 + u;
            const int hidx = j * 16 + nl;
            #pragma unroll
            for (int r = 0; r < 8; ++r) {
                const int bl = r + 8 * half;
                hl[bl * H_DIM + hidx] = (bf16)hnew[u][r];
            }
        }
        __syncthreads();   // hl updated for next step
    }
}

// ---------------------------------------------------------------------------
// Kernel: local[b,k,c] forward/backward section differences from pad_output
// ---------------------------------------------------------------------------
__global__ void k_local_diff(const bf16* __restrict__ po,
                             const int* __restrict__ begin,
                             const int* __restrict__ end,
                             float* __restrict__ local) {
    int idx = blockIdx.x * blockDim.x + threadIdx.x;
    if (idx >= B_SZ * K_SEC * D2H) return;
    int c = idx & (D2H - 1);
    int k = (idx >> 9) & (K_SEC - 1);
    int b = idx >> 15;
    int te = end[b * K_SEC + k];
    int tb = begin[b * K_SEC + k];
    float pe = (float)po[((size_t)b * (S_LEN + 1) + te) * D2H + c];
    float pb = (float)po[((size_t)b * (S_LEN + 1) + tb) * D2H + c];
    local[idx] = (c < H_DIM) ? (pe - pb) : (pb - pe);
}

// ---------------------------------------------------------------------------
// Kernel: lcr gather. section_indicator is one-hot [B,S,K]; find the hot k,
// copy local[b,k,:] into mlp_in cols [512,1024) as bf16. One block per (s,b).
// ---------------------------------------------------------------------------
__global__ __launch_bounds__(256) void k_lcr_gather(const float* __restrict__ ind,
                                                    const float* __restrict__ local,
                                                    bf16* __restrict__ mlp_in) {
    const int sb = blockIdx.x;
    const int s = sb >> 6;
    const int b = sb & (B_SZ - 1);
    __shared__ int kk;
    if (threadIdx.x == 0) {
        const float* p = ind + (size_t)b * S_LEN * K_SEC + (size_t)s * K_SEC;
        int kv = 0;
        for (int k = 0; k < K_SEC; ++k) {
            if (p[k] > 0.5f) { kv = k; break; }
        }
        kk = kv;
    }
    __syncthreads();
    const float* lp = local + ((size_t)b * K_SEC + kk) * D2H;
    bf16* op = mlp_in + (size_t)sb * D4H + D2H;
    for (int c = threadIdx.x; c < D2H; c += 256) op[c] = (bf16)lp[c];
}

// ---------------------------------------------------------------------------
// Kernel: fused MLP.  out[r] = W2 . relu(W1 . mlp_in[r] + b1) + b2
//   A : bf16 [65536, 1024], W1 : bf16 [512, 1024], W2/b1/b2 fp32.
// Block = 256 threads (8 waves), one 16-row tile per block; A slab staged via
// async-to-LDS. One A fragment per k-step feeds the wave's 4 column tiles.
// After ReLU, each lane dots its column with W2; shfl_xor reduction over the
// 16 N-lanes; deterministic per-wave partials in LDS.
// ---------------------------------------------------------------------------
__global__ __launch_bounds__(256) void k_mlp_fused(const bf16* __restrict__ A,
                                                   const bf16* __restrict__ W1,
                                                   const float* __restrict__ b1,
                                                   const float* __restrict__ W2,
                                                   const float* __restrict__ b2,
                                                   float* __restrict__ out) {
    __shared__ bf16 As[16 * D4H];          // 32 KB
    __shared__ float accw[8][16];

    const int rb = blockIdx.x * 16;
    for (int c = threadIdx.x; c < 16 * D4H / 8; c += 256) {
        int r  = c >> 7;                    // 128 chunks of 8 bf16 per row
        int kc = (c & 127) << 3;
        async_copy16(As + r * D4H + kc, A + ((size_t)(rb + r)) * D4H + kc);
    }
    async_wait_all();
    __syncthreads();

    const int wave = threadIdx.x >> 5;
    const int lane = threadIdx.x & 31;
    const int half = lane >> 4, nl = lane & 15;

    v8f acc[4] = {ZERO8, ZERO8, ZERO8, ZERO8};
    #pragma unroll 4
    for (int k = 0; k < D4H; k += 32) {
        v16bf a = load_frag(As, D4H, 0, k);
        #pragma unroll
        for (int ct = 0; ct < 4; ++ct) {
            const int cbase = (wave * 4 + ct) * 16;
            acc[ct] = wmma_bf16(a, load_frag(W1, D4H, cbase, k), acc[ct]);
        }
    }

    float part[8] = {0.f, 0.f, 0.f, 0.f, 0.f, 0.f, 0.f, 0.f};
    #pragma unroll
    for (int ct = 0; ct < 4; ++ct) {
        const int col = (wave * 4 + ct) * 16 + nl;
        const float w2v = W2[col], b1v = b1[col];
        #pragma unroll
        for (int r = 0; r < 8; ++r) {
            float h = acc[ct][r] + b1v;
            h = h > 0.f ? h : 0.f;
            part[r] += h * w2v;
        }
    }
    // reduce over the 16 lanes holding different columns (stays within halves)
    for (int m = 1; m < 16; m <<= 1)
        for (int r = 0; r < 8; ++r) part[r] += __shfl_xor(part[r], m, 32);
    if (nl == 0)
        for (int r = 0; r < 8; ++r) accw[wave][r + 8 * half] = part[r];
    __syncthreads();
    if (threadIdx.x < 16) {
        float sum = b2[0];
        for (int w = 0; w < 8; ++w) sum += accw[w][threadIdx.x];
        out[rb + threadIdx.x] = sum;
    }
}

// ---------------------------------------------------------------------------
// Workspace layout (bytes, all 256-aligned)
// ---------------------------------------------------------------------------
#define OFF_XBF    ((size_t)0)                       // x bf16        67,108,864
#define OFF_WIHF   ((size_t)67108864)                // 768x512 bf16     786,432
#define OFF_WHHF   ((size_t)67895296)                // 768x256 bf16     393,216
#define OFF_WIHB   ((size_t)68288512)                //                  786,432
#define OFF_WHHB   ((size_t)69074944)                //                  393,216
#define OFF_W1BF   ((size_t)69468160)                // 512x1024 bf16  1,048,576
#define OFF_XIF    ((size_t)70516736)                // bf16 65536x768 100,663,296
#define OFF_XIB    ((size_t)171180032)               //               100,663,296
#define OFF_PO     ((size_t)271843328)               // bf16 64x1025x512 67,174,400
#define OFF_MLPIN  ((size_t)339017728)               // bf16 65536x1024 134,217,728
#define OFF_LOCAL  ((size_t)473235456)               // f32 64x64x512    8,388,608

extern "C" void kernel_launch(void* const* d_in, const int* in_sizes, int n_in,
                              void* d_out, int out_size, void* d_ws, size_t ws_size,
                              hipStream_t stream) {
    const float* inputs = (const float*)d_in[0];
    // d_in[1] = input_length (full-length, unused)
    const float* ind    = (const float*)d_in[2];
    const int*   begin  = (const int*)d_in[3];
    const int*   endp   = (const int*)d_in[4];
    const float* W_ih_f = (const float*)d_in[5];
    const float* W_hh_f = (const float*)d_in[6];
    const float* b_ih_f = (const float*)d_in[7];
    const float* b_hh_f = (const float*)d_in[8];
    const float* W_ih_b = (const float*)d_in[9];
    const float* W_hh_b = (const float*)d_in[10];
    const float* b_ih_b = (const float*)d_in[11];
    const float* b_hh_b = (const float*)d_in[12];
    const float* W1     = (const float*)d_in[13];
    const float* b1     = (const float*)d_in[14];
    const float* W2     = (const float*)d_in[15];
    const float* b2     = (const float*)d_in[16];
    float* out = (float*)d_out;

    char* ws = (char*)d_ws;
    bf16* xbf   = (bf16*)(ws + OFF_XBF);
    bf16* wihf  = (bf16*)(ws + OFF_WIHF);
    bf16* whhf  = (bf16*)(ws + OFF_WHHF);
    bf16* wihb  = (bf16*)(ws + OFF_WIHB);
    bf16* whhb  = (bf16*)(ws + OFF_WHHB);
    bf16* w1bf  = (bf16*)(ws + OFF_W1BF);
    bf16* xif   = (bf16*)(ws + OFF_XIF);
    bf16* xib   = (bf16*)(ws + OFF_XIB);
    bf16* po    = (bf16*)(ws + OFF_PO);
    bf16* mlpin = (bf16*)(ws + OFF_MLPIN);
    float* local = (float*)(ws + OFF_LOCAL);

    // 1) conversions to bf16 (4-wide)
    k_f32_to_bf16<<<4096, 256, 0, stream>>>(inputs, xbf, (size_t)ROWS * I_DIM / 4);
    k_f32_to_bf16<<<384,  256, 0, stream>>>(W_ih_f, wihf, (size_t)G3H * I_DIM / 4);
    k_f32_to_bf16<<<192,  256, 0, stream>>>(W_hh_f, whhf, (size_t)G3H * H_DIM / 4);
    k_f32_to_bf16<<<384,  256, 0, stream>>>(W_ih_b, wihb, (size_t)G3H * I_DIM / 4);
    k_f32_to_bf16<<<192,  256, 0, stream>>>(W_hh_b, whhb, (size_t)G3H * H_DIM / 4);
    k_f32_to_bf16<<<512,  256, 0, stream>>>(W1, w1bf, (size_t)MLP_D * D4H / 4);
    k_zero_po_row0<<<(B_SZ * D2H + 255) / 256, 256, 0, stream>>>(po);

    // 2) input projections (WMMA GEMM): xi = x @ W_ih^T + b_ih
    {
        dim3 grid(G3H / 64, ROWS / 128);
        k_gemm_xi<<<grid, 256, 0, stream>>>(xbf, wihf, b_ih_f, xif);
        k_gemm_xi<<<grid, 256, 0, stream>>>(xbf, wihb, b_ih_b, xib);
    }

    // 3) bi-GRU recurrence (batch-parallel persistent scan)
    {
        dim3 grid(2, B_SZ / 16);
        k_gru_scan<<<grid, 256, 0, stream>>>(xif, xib, whhf, whhb,
                                             b_hh_f, b_hh_b, po, mlpin);
    }

    // 4) section differences + lcr gather into mlp_in cols [512,1024)
    k_local_diff<<<(B_SZ * K_SEC * D2H + 255) / 256, 256, 0, stream>>>(po, begin, endp, local);
    k_lcr_gather<<<ROWS, 256, 0, stream>>>(ind, local, mlpin);

    // 5) fused MLP (WMMA GEMM + ReLU + GEMV + bias)
    k_mlp_fused<<<ROWS / 16, 256, 0, stream>>>(mlpin, w1bf, b1, W2, b2, out);
}
